// LinearEncoder_12472585028062
// MI455X (gfx1250) — matrix operand order repaired and checked by reading, not compiled, not run on hardware
//
#include <hip/hip_runtime.h>

#define DIN  512
#define DOUT 512

typedef __attribute__((ext_vector_type(16))) __bf16 v16bf;
typedef __attribute__((ext_vector_type(8)))  float  v8f;

// ---------------------------------------------------------------------------
// dropout + fp32 -> bf16 convert
__device__ __forceinline__ __bf16 dc(float x, float m) {
    float v = (m >= 0.1f) ? x * (1.0f / 0.9f) : 0.0f;
    return (__bf16)v;
}

// ---------------------------------------------------------------------------
__global__ __launch_bounds__(256) void k_init_deg(float* __restrict__ deg, int n) {
    int i = blockIdx.x * 256 + threadIdx.x;
    if (i < n) deg[i] = 1.0f;   // self loop
}

__global__ __launch_bounds__(256) void k_count_deg(const int* __restrict__ ei,
                                                   float* __restrict__ deg, int E) {
    int e = blockIdx.x * 256 + threadIdx.x;
    if (e < E) unsafeAtomicAdd(&deg[ei[E + e]], 1.0f);
}

__global__ __launch_bounds__(256) void k_dinv(float* __restrict__ deg, int n) {
    int i = blockIdx.x * 256 + threadIdx.x;
    if (i < n) deg[i] = rsqrtf(deg[i]);
}

// ---------------------------------------------------------------------------
// Pre-shuffle W [DIN,DOUT] fp32 row-major into per-lane bf16 B-fragments for
// v_wmma_f32_16x16x32_bf16 (B is 32x16 KxN; lanes 0-15: K0-7/16-23 of col=lane,
// lanes 16-31: K8-15/24-31; 16 bf16 contiguous per lane = 32B).
// wfrag index: ((kt*32 + nt)*32 + lane)*16 + j ,  k = kt*32 + (lane>>4)*8 + (j&7) + (j>>3)*16
__global__ __launch_bounds__(256) void k_wfrag(const float* __restrict__ W,
                                               __bf16* __restrict__ wfrag) {
    int tid  = blockIdx.x * 256 + threadIdx.x;   // 0 .. 512*512-1
    int j    = tid & 15;
    int lane = (tid >> 4) & 31;
    int nt   = (tid >> 9) & 31;
    int kt   = tid >> 14;
    int col  = nt * 16 + (lane & 15);
    int k    = kt * 32 + (lane >> 4) * 8 + (j & 7) + (j >> 3) * 16;
    wfrag[tid] = (__bf16)W[k * DOUT + col];
}

// ---------------------------------------------------------------------------
// Pipelined fused GEMM: g = dinv .* ((x .* dropout) @ W); also seeds out = g.
struct RawA { float4 t0, t1, t2, t3, u0, u1, u2, u3; };
struct BSet { v16bf b0, b1, b2, b3; };

__device__ __forceinline__ void load_rawA(RawA& r, const float* __restrict__ xrow,
                                          const float* __restrict__ mrow, int k0) {
    const float4* xp = (const float4*)(xrow + k0);
    const float4* mp = (const float4*)(mrow + k0);
    r.t0 = xp[0]; r.t1 = xp[1]; r.t2 = xp[4]; r.t3 = xp[5];
    r.u0 = mp[0]; r.u1 = mp[1]; r.u2 = mp[4]; r.u3 = mp[5];
}

__device__ __forceinline__ void load_bset(BSet& b, const __bf16* __restrict__ wfrag,
                                          int kt, int ntBase, int lane) {
    const size_t wb = (((size_t)kt * 32 + ntBase) * 32 + lane) * 16;
    b.b0 = *(const v16bf*)(wfrag + wb);
    b.b1 = *(const v16bf*)(wfrag + wb + 512);
    b.b2 = *(const v16bf*)(wfrag + wb + 1024);
    b.b3 = *(const v16bf*)(wfrag + wb + 1536);
}

__device__ __forceinline__ v16bf cvt_a(const RawA& r) {
    v16bf a;
    a[0]  = dc(r.t0.x, r.u0.x); a[1]  = dc(r.t0.y, r.u0.y);
    a[2]  = dc(r.t0.z, r.u0.z); a[3]  = dc(r.t0.w, r.u0.w);
    a[4]  = dc(r.t1.x, r.u1.x); a[5]  = dc(r.t1.y, r.u1.y);
    a[6]  = dc(r.t1.z, r.u1.z); a[7]  = dc(r.t1.w, r.u1.w);
    a[8]  = dc(r.t2.x, r.u2.x); a[9]  = dc(r.t2.y, r.u2.y);
    a[10] = dc(r.t2.z, r.u2.z); a[11] = dc(r.t2.w, r.u2.w);
    a[12] = dc(r.t3.x, r.u3.x); a[13] = dc(r.t3.y, r.u3.y);
    a[14] = dc(r.t3.z, r.u3.z); a[15] = dc(r.t3.w, r.u3.w);
    return a;
}

#define WMMA4(A, B)                                                              \
    acc0 = __builtin_amdgcn_wmma_f32_16x16x32_bf16(false, A, false, (B).b0,      \
                                                   (short)0, acc0, false, false);\
    acc1 = __builtin_amdgcn_wmma_f32_16x16x32_bf16(false, A, false, (B).b1,      \
                                                   (short)0, acc1, false, false);\
    acc2 = __builtin_amdgcn_wmma_f32_16x16x32_bf16(false, A, false, (B).b2,      \
                                                   (short)0, acc2, false, false);\
    acc3 = __builtin_amdgcn_wmma_f32_16x16x32_bf16(false, A, false, (B).b3,      \
                                                   (short)0, acc3, false, false)

__global__ __launch_bounds__(256)
void k_gcn_gemm(const float* __restrict__ x,
                const float* __restrict__ mask,
                const __bf16* __restrict__ wfrag,
                const float* __restrict__ dinv,
                float* __restrict__ g,
                float* __restrict__ out,
                int nrows) {
    const int lane = threadIdx.x & 31;
    const int wave = threadIdx.x >> 5;
    const int hf   = lane >> 4;
    const int l16  = lane & 15;

    const int rowBase = blockIdx.x * 128 + wave * 16;
    const int ntBase  = blockIdx.y * 4;
    const int colBase = ntBase * 16;

    // A source row; clamp so all 32 lanes stay active through WMMA
    int rowA = rowBase + l16;
    if (rowA >= nrows) rowA = nrows - 1;
    const float* xrow = x    + (size_t)rowA * DIN;
    const float* mrow = mask + (size_t)rowA * DIN;
    const int kb = hf * 8;  // ISA A layout: lanes0-15 K0-7/16-23, lanes16-31 K8-15/24-31

    v8f acc0 = {}, acc1 = {}, acc2 = {}, acc3 = {};

    RawA r;
    BSet B0, B1;
    v16bf a0, a1;

    // prologue: stage kt=0
    load_rawA(r, xrow, mrow, kb);
    load_bset(B0, wfrag, 0, ntBase, lane);
    a0 = cvt_a(r);

#pragma unroll
    for (int kt = 0; kt < DIN / 32; kt += 2) {
        // issue kt+1 loads, then run kt's WMMAs while they fly
        load_rawA(r, xrow, mrow, (kt + 1) * 32 + kb);
        load_bset(B1, wfrag, kt + 1, ntBase, lane);
        WMMA4(a0, B0);
        a1 = cvt_a(r);

        if (kt + 2 < DIN / 32) {
            load_rawA(r, xrow, mrow, (kt + 2) * 32 + kb);
            load_bset(B0, wfrag, kt + 2, ntBase, lane);
        }
        WMMA4(a1, B1);
        if (kt + 2 < DIN / 32) a0 = cvt_a(r);
    }

    // C/D layout: VGPR j -> row M = hf*8 + j, col N = colBase + tile*16 + l16
    const bool fullBlock = (rowBase - wave * 16 + 128) <= nrows;  // uniform
    if (fullBlock) {
#pragma unroll
        for (int j = 0; j < 8; ++j) {
            const int row = rowBase + hf * 8 + j;
            const float dv = dinv[row];
            const size_t o = (size_t)row * DOUT + colBase + l16;
            float v0 = acc0[j] * dv, v1 = acc1[j] * dv;
            float v2 = acc2[j] * dv, v3 = acc3[j] * dv;
            g[o]        = v0;  g[o + 16]   = v1;
            g[o + 32]   = v2;  g[o + 48]   = v3;
            out[o]      = v0;  out[o + 16] = v1;
            out[o + 32] = v2;  out[o + 48] = v3;
        }
    } else {
#pragma unroll
        for (int j = 0; j < 8; ++j) {
            const int row = rowBase + hf * 8 + j;
            if (row < nrows) {
                const float dv = dinv[row];
                const size_t o = (size_t)row * DOUT + colBase + l16;
                float v0 = acc0[j] * dv, v1 = acc1[j] * dv;
                float v2 = acc2[j] * dv, v3 = acc3[j] * dv;
                g[o]        = v0;  g[o + 16]   = v1;
                g[o + 32]   = v2;  g[o + 48]   = v3;
                out[o]      = v0;  out[o + 16] = v1;
                out[o + 32] = v2;  out[o + 48] = v3;
            }
        }
    }
}

// ---------------------------------------------------------------------------
// out[dst] += g[src] per edge; 128 threads/edge, float4 gather + 4 f32 atomics
__global__ __launch_bounds__(256)
void k_scatter(const int* __restrict__ ei, const float* __restrict__ g,
               float* __restrict__ out, int E) {
    const int tid = blockIdx.x * 256 + threadIdx.x;
    const int e = tid >> 7;
    if (e >= E) return;
    const int f = (tid & 127) << 2;
    const int r = ei[e];
    const int c = ei[E + e];
    const float4 gv = *(const float4*)(g + (size_t)r * DOUT + f);
    float* op = out + (size_t)c * DOUT + f;
    unsafeAtomicAdd(op + 0, gv.x);
    unsafeAtomicAdd(op + 1, gv.y);
    unsafeAtomicAdd(op + 2, gv.z);
    unsafeAtomicAdd(op + 3, gv.w);
}

// ---------------------------------------------------------------------------
// out = dinv[row] * out + b   (in place)
__global__ __launch_bounds__(256)
void k_finish(float* __restrict__ out, const float* __restrict__ dinv,
              const float* __restrict__ b, int n) {
    const int tid = blockIdx.x * 256 + threadIdx.x;
    const int i = tid >> 7;
    if (i >= n) return;
    const int f = (tid & 127) << 2;
    const float dv = dinv[i];
    const float4 bv = *(const float4*)(b + f);
    float4 ov = *(const float4*)(out + (size_t)i * DOUT + f);
    ov.x = ov.x * dv + bv.x;
    ov.y = ov.y * dv + bv.y;
    ov.z = ov.z * dv + bv.z;
    ov.w = ov.w * dv + bv.w;
    *(float4*)(out + (size_t)i * DOUT + f) = ov;
}

// ---------------------------------------------------------------------------
extern "C" void kernel_launch(void* const* d_in, const int* in_sizes, int n_in,
                              void* d_out, int out_size, void* d_ws, size_t ws_size,
                              hipStream_t stream) {
    const float* x    = (const float*)d_in[0];
    const int*   ei   = (const int*)d_in[1];    // edge_index [2,E] (int32)
    const float* W    = (const float*)d_in[2];
    const float* bias = (const float*)d_in[3];
    const float* mask = (const float*)d_in[4];
    float* out = (float*)d_out;

    const int N = in_sizes[0] / DIN;  // 100000
    const int E = in_sizes[1] / 2;    // 400000

    // workspace layout
    char* ws = (char*)d_ws;
    float*  deg   = (float*)ws;                       // N f32 (becomes dinv)
    __bf16* wfrag = (__bf16*)(ws + (512u << 10));     // 512KB bf16 fragments
    float*  g     = (float*)(ws + (1u << 20));        // N x DOUT f32

    k_init_deg<<<(N + 255) / 256, 256, 0, stream>>>(deg, N);
    k_count_deg<<<(E + 255) / 256, 256, 0, stream>>>(ei, deg, E);
    k_dinv<<<(N + 255) / 256, 256, 0, stream>>>(deg, N);
    k_wfrag<<<(DIN * DOUT) / 256, 256, 0, stream>>>(W, wfrag);

    dim3 ggrid((N + 127) / 128, DOUT / 64);
    k_gcn_gemm<<<ggrid, 256, 0, stream>>>(x, mask, wfrag, deg, g, out, N);

    long long sthreads = (long long)E * 128;
    k_scatter<<<(unsigned)((sthreads + 255) / 256), 256, 0, stream>>>(ei, g, out, E);

    long long fthreads = (long long)N * 128;
    k_finish<<<(unsigned)((fthreads + 255) / 256), 256, 0, stream>>>(out, deg, bias, N);
}
